// ScaledDotProductAttention_30614526885917
// MI455X (gfx1250) — matrix operand order repaired and checked by reading, not compiled, not run on hardware
//
#include <hip/hip_runtime.h>
#include <stdint.h>

// ---------------------------------------------------------------------------
// Problem constants (fixed by the reference: B=8, Lq=Lk=2048, d=dv=64, C=5)
// ---------------------------------------------------------------------------
#define B_    8
#define LQ    2048
#define LK    2048
#define D_    64
#define NC    5
#define INV_T 0.125f   // 1/TEMPERATURE, folded into K pre-conversion

typedef _Float16 v16h __attribute__((ext_vector_type(16)));
typedef float    v8f  __attribute__((ext_vector_type(8)));

__device__ __forceinline__ float shfl16(float x) {
  return __shfl_xor(x, 16, 32);
}

// ---------------------------------------------------------------------------
// Kernel 0: zero the class accumulators (graph-capture-safe, no memset)
// ---------------------------------------------------------------------------
__global__ void zero_ws_kernel(float* ws, int n) {
  int i = blockIdx.x * blockDim.x + threadIdx.x;
  if (i < n) ws[i] = 0.0f;
}

// ---------------------------------------------------------------------------
// Kernel P1: pre-swizzle K into f16 A-operand layout, scaled by 1/T, with a
// key-row PERMUTATION baked in so that the Sᵀ result lands lane-aligned for
// the PV B-operand (no cross-lane exchange needed in the hot loop):
//   32-key block kblk, tile T = kt&1, A-row M = lid maps to global key
//     n = kblk*32 + (lid<8 ? T*8 + lid : 16 + T*8 + (lid-8))
//   => after WMMA (C-layout M = r + 8h), lane-half h owns keys h*16+0..15
//      in B-operand element order.
// Layout: Kswz[ ((b*128 + kt)*2 + c)*32 + lane ][16 halves]
//   element j -> feat:  c*32 + h*16 + j
// Each thread emits one 8-half group (16B store, contiguous 8-float read).
// ---------------------------------------------------------------------------
__global__ __launch_bounds__(256)
void prep_k_kernel(const float* __restrict__ Km, _Float16* __restrict__ Kswz) {
  const int idx  = blockIdx.x * blockDim.x + threadIdx.x;   // 131072 threads
  const int g    = idx & 1;
  const int lane = (idx >> 1) & 31;
  const int c    = (idx >> 6) & 1;
  const int kt   = (idx >> 7) & 127;
  const int b    = idx >> 14;
  const int h = lane >> 4, lid = lane & 15;
  const int T    = kt & 1;
  const int kblk = kt >> 1;
  const int n    = kblk * 32 + ((lid < 8) ? (T * 8 + lid)
                                          : (16 + T * 8 + (lid - 8)));
  const int feat = c * 32 + h * 16 + g * 8;
  const float* src = Km + ((size_t)(b * LK + n)) * D_ + feat;
  _Float16* dst = Kswz + (size_t)idx * 8;
#pragma unroll
  for (int j = 0; j < 8; ++j) dst[j] = (_Float16)(src[j] * INV_T);
}

// ---------------------------------------------------------------------------
// Kernel P2: pre-swizzle Vᵀ into f16 A-operand layout (contraction index =
// plain global key, matching the B-operand element order above).
// Layout: Vswz[ ((b*64 + kblk)*4 + t)*32 + lane ][16 halves]
//   lane = (h,lid): d row = t*16+lid, element j -> key:
//     kblk*32 + h*16 + j
// ---------------------------------------------------------------------------
__global__ __launch_bounds__(256)
void prep_v_kernel(const float* __restrict__ Vm, _Float16* __restrict__ Vswz) {
  const int idx  = blockIdx.x * blockDim.x + threadIdx.x;   // 131072 threads
  const int g    = idx & 1;
  const int lane = (idx >> 1) & 31;
  const int t    = (idx >> 6) & 3;
  const int kblk = (idx >> 8) & 63;
  const int b    = idx >> 14;
  const int h = lane >> 4, lid = lane & 15;
  const int d  = t * 16 + lid;
  const int n0 = kblk * 32 + h * 16 + g * 8;
  const float* src = Vm + ((size_t)(b * LK + n0)) * D_ + d;
  _Float16* dst = Vswz + (size_t)idx * 8;
#pragma unroll
  for (int j = 0; j < 8; ++j) dst[j] = (_Float16)src[(size_t)j * D_];
}

// ---------------------------------------------------------------------------
// Kernel 1: flash attention (Sᵀ/Oᵀ formulation on WMMA) + class accumulation.
// One wave = one 16-query tile. Block = 256 threads = 8 waves = 128 queries.
// Grid = B * (LQ/16/8) = 128 blocks.
//
// Sᵀ WMMA:  D[key][query] = Ktile(16x32) x Qop(32x16)   (f16 in, f32 acc)
// Oᵀ WMMA:  D[d][query]  += Vᵀtile(16x32keys) x Pop(32keys x 16 queries)
//
// Lane pair (lid, lid+16) co-owns query `lid`; with the permuted K swizzle,
// each lane's 16 exp'd scores ARE the PV B-operand elements in order, so the
// loop body is: 8 aligned v16h loads, 8 v_wmma, per-lane softmax with two
// xor-16 reductions, and 8 packed f16 conversions. No LDS, no exchanges.
// ---------------------------------------------------------------------------
__global__ __launch_bounds__(256)
void attn_class_kernel(const float* __restrict__ Qm,
                       const _Float16* __restrict__ Kswz,
                       const _Float16* __restrict__ Vswz,
                       const int64_t* __restrict__ Y,
                       float* __restrict__ Scls,   // [B][NC][64]
                       float* __restrict__ Cnt)    // [B][NC]
{
  const int wave = threadIdx.x >> 5;
  const int lane = threadIdx.x & 31;
  const int h    = lane >> 4;     // half-wave: selects K-range of operands
  const int lid  = lane & 15;

  const int blocksPerBatch = (LQ / 16) / 8;               // 16
  const int b  = blockIdx.x / blocksPerBatch;
  const int qt = (blockIdx.x % blocksPerBatch) * 8 + wave;
  const int q0 = qt * 16;

  // --- Q as B-operand (lane = query column lid; element j -> K = h*16 + j) --
  const float* qrow = Qm + ((size_t)(b * LQ + q0 + lid)) * D_;
  v16h qb[2];
#pragma unroll
  for (int c = 0; c < 2; ++c) {
    const float* p = qrow + c * 32 + h * 16;
#pragma unroll
    for (int j = 0; j < 16; ++j) qb[c][j] = (_Float16)p[j];
  }

  float mstate = -3.0e38f;   // running row max (of pre-scaled logits)
  float lstate = 0.0f;       // running row sum
  v8f acc[4] = {v8f{}, v8f{}, v8f{}, v8f{}};   // Oᵀ accum: 4 d-tiles x 16 q
  const v8f zero = {};

  // Per-iteration operand strides (in halves): c-stride 512, kt-stride 1024.
  const size_t kLane = (size_t)lane * 16;
  for (int kblk = 0; kblk < LK / 32; ++kblk) {
    const _Float16* kp = Kswz + ((size_t)(b * 128 + kblk * 2)) * 1024 + kLane;
    const v16h a00 = *(const v16h*)(kp);
    const v16h a01 = *(const v16h*)(kp + 512);
    const v16h a10 = *(const v16h*)(kp + 1024);
    const v16h a11 = *(const v16h*)(kp + 1536);

    // ---- Sᵀ tiles: rows = permuted keys, cols = queries (logits pre-/T) --
    v8f s0 = __builtin_amdgcn_wmma_f32_16x16x32_f16(false, a00, false, qb[0],
                                                    (short)0, zero, false, false);
    s0     = __builtin_amdgcn_wmma_f32_16x16x32_f16(false, a01, false, qb[1],
                                                    (short)0, s0, false, false);
    v8f s1 = __builtin_amdgcn_wmma_f32_16x16x32_f16(false, a10, false, qb[0],
                                                    (short)0, zero, false, false);
    s1     = __builtin_amdgcn_wmma_f32_16x16x32_f16(false, a11, false, qb[1],
                                                    (short)0, s1, false, false);

    // ---- online softmax for this lane's query (16 of 32 keys per lane) ---
    // With the baked permutation, s0[r] = key h*16+r, s1[r] = key h*16+8+r.
    float mloc = s0[0];
#pragma unroll
    for (int r = 1; r < 8; ++r) mloc = fmaxf(mloc, s0[r]);
#pragma unroll
    for (int r = 0; r < 8; ++r) mloc = fmaxf(mloc, s1[r]);
    mloc = fmaxf(mloc, shfl16(mloc));
    const float mnew = fmaxf(mstate, mloc);
    const float corr = __expf(mstate - mnew);

    v16h pb;           // PV B-operand: element j -> key h*16 + j  (direct!)
    float ls = 0.0f;
#pragma unroll
    for (int r = 0; r < 8; ++r) {
      const float e = __expf(s0[r] - mnew);
      ls += e;
      pb[r] = (_Float16)e;
    }
#pragma unroll
    for (int r = 0; r < 8; ++r) {
      const float e = __expf(s1[r] - mnew);
      ls += e;
      pb[8 + r] = (_Float16)e;
    }
    ls += shfl16(ls);
    lstate = lstate * corr + ls;
    mstate = mnew;

#pragma unroll
    for (int t = 0; t < 4; ++t)
#pragma unroll
      for (int r = 0; r < 8; ++r) acc[t][r] *= corr;

    // ---- Oᵀ += Vᵀ x P : A rows = output dims, K = 32 keys ----------------
    const _Float16* vp = Vswz + ((size_t)(b * 64 + kblk)) * 2048 + kLane;
#pragma unroll
    for (int t = 0; t < 4; ++t) {
      const v16h va = *(const v16h*)(vp + (size_t)t * 512);
      acc[t] = __builtin_amdgcn_wmma_f32_16x16x32_f16(false, va, false, pb,
                                                      (short)0, acc[t], false, false);
    }
  }

  // ---- finalize: normalize by row sum, accumulate into class buckets -----
  const float invl = 1.0f / lstate;
  const int q   = q0 + lid;
  const int cls = (int)Y[(size_t)b * LQ + q];
  float* srow = Scls + ((size_t)(b * NC + cls)) * D_;
#pragma unroll
  for (int t = 0; t < 4; ++t)
#pragma unroll
    for (int r = 0; r < 8; ++r) {
      const int d = t * 16 + r + 8 * h;      // C-layout: VGPR r -> M = r + 8*half
      atomicAdd(&srow[d], acc[t][r] * invl);
    }
  if (h == 0) atomicAdd(&Cnt[b * NC + cls], 1.0f);
}

// ---------------------------------------------------------------------------
// Kernel 2: out[b,q,:] = Scls[b, Y[b,q], :] / Cnt[b, Y[b,q]]
// ---------------------------------------------------------------------------
__global__ __launch_bounds__(256)
void bcast_kernel(const int64_t* __restrict__ Y,
                  const float* __restrict__ Scls,
                  const float* __restrict__ Cnt,
                  float* __restrict__ out)
{
  const int idx = blockIdx.x * blockDim.x + threadIdx.x;
  if (idx >= B_ * LQ * D_) return;
  const int d = idx & (D_ - 1);
  const int q = (idx >> 6) & (LQ - 1);
  const int b = idx >> 17;                 // / (LQ*D_) with LQ*D_ = 131072
  const int cls = (int)Y[(size_t)b * LQ + q];
  out[idx] = Scls[(size_t)(b * NC + cls) * D_ + d] / Cnt[b * NC + cls];
}

// ---------------------------------------------------------------------------
extern "C" void kernel_launch(void* const* d_in, const int* in_sizes, int n_in,
                              void* d_out, int out_size, void* d_ws, size_t ws_size,
                              hipStream_t stream) {
  const float*   Qm = (const float*)d_in[0];
  const float*   Km = (const float*)d_in[1];
  const float*   Vm = (const float*)d_in[2];
  const int64_t* Y  = (const int64_t*)d_in[3];

  // ---- workspace layout -------------------------------------------------
  // [0)            Scls : B*NC*64 floats   (10240 B)
  // [.. +160 B)    Cnt  : B*NC floats      (160 B)   -> 10400 B, 32B-aligned
  // [.. +2 MiB)    Kswz : B*128*2*32*16 halves (pre-swizzled, pre-scaled,
  //                       key rows permuted for lane-aligned P)
  // [.. +2 MiB)    Vswz : B*64*4*32*16 halves
  float* Scls = (float*)d_ws;
  float* Cnt  = Scls + B_ * NC * D_;
  const int nws = B_ * NC * D_ + B_ * NC;         // 2600 floats
  _Float16* Kswz = (_Float16*)((char*)d_ws + 10400);
  _Float16* Vswz = Kswz + (size_t)B_ * 128 * 2 * 32 * 16;   // +1,048,576 halves

  zero_ws_kernel<<<(nws + 255) / 256, 256, 0, stream>>>(Scls, nws);

  prep_k_kernel<<<131072 / 256, 256, 0, stream>>>(Km, Kswz);
  prep_v_kernel<<<131072 / 256, 256, 0, stream>>>(Vm, Vswz);

  const int grid1 = B_ * ((LQ / 16) / 8);     // 128 blocks x 256 threads
  attn_class_kernel<<<grid1, 256, 0, stream>>>(Qm, Kswz, Vswz, Y, Scls, Cnt);

  const int n = B_ * LQ * D_;
  bcast_kernel<<<(n + 255) / 256, 256, 0, stream>>>(Y, Scls, Cnt, (float*)d_out);
}
